// AdaptMultiheadAttention_16947940950046
// MI455X (gfx1250) — matrix operand order repaired and checked by reading, not compiled, not run on hardware
//
#include <hip/hip_runtime.h>
#include <math.h>
#include <stdint.h>

typedef _Float16 f16;
typedef __attribute__((ext_vector_type(16))) _Float16 v16h;
typedef __attribute__((ext_vector_type(8)))  float    v8f;

// Problem constants (from reference)
constexpr int Bc = 2, Tt = 2048, Hh = 16, Cc = 64, Ww = 1024;
constexpr float SCALE2 = 0.125f; // (C^-0.25)^2 = 1/sqrt(64), pre-folded into Q

// ---------------------------------------------------------------------------
// WMMA helpers (wave32, 16x16x32 f16 -> f32)
// ---------------------------------------------------------------------------
__device__ __forceinline__ v8f wmma_f32(v16h a, v16h b, v8f c) {
  return __builtin_amdgcn_wmma_f32_16x16x32_f16(
      /*neg_a=*/false, a, /*neg_b=*/false, b,
      /*c_mod=*/(short)0, c, /*reuse_a=*/false, /*reuse_b=*/false);
}

// A operand: 16x32 (MxK) tile, row-major in LDS with stride `ld` (f16 elems).
__device__ __forceinline__ v16h load_a_frag(const f16* base, int ld) {
  const int lane = threadIdx.x & 31;
  const int m = lane & 15, hf = lane >> 4;
  const f16* r = base + m * ld;
  v16h a;
#pragma unroll
  for (int j = 0; j < 8; ++j) {
    int k = hf * 16 + j * 2;
    a[2 * j]     = r[k];
    a[2 * j + 1] = r[k + 1];
  }
  return a;
}

// B operand: 32x16 (KxN). LDS tile stored transposed [n][k] with stride `ld`.
__device__ __forceinline__ v16h load_b_frag(const f16* base, int ld) {
  const int lane = threadIdx.x & 31;
  const int n = lane & 15, hf = lane >> 4;
  const f16* r = base + n * ld + hf * 16;
  v16h b;
#pragma unroll
  for (int j = 0; j < 16; ++j) b[j] = r[j];
  return b;
}

// Async copy 16B global -> LDS (ASYNCcnt path), GVS addressing mode.
__device__ __forceinline__ void async_copy_b128(uint32_t lds_off,
                                                uint64_t gbase,
                                                uint32_t goff) {
  asm volatile("global_load_async_to_lds_b128 %0, %1, %2 offset:0"
               :: "v"(lds_off), "v"(goff), "s"(gbase)
               : "memory");
}
__device__ __forceinline__ void async_wait0() {
  asm volatile("s_wait_asynccnt 0x0" ::: "memory");
}
__device__ __forceinline__ uint32_t lds_off_of(const void* p) {
  return (uint32_t)(uintptr_t)p;  // low 32 bits of generic addr = LDS offset
}

// ---------------------------------------------------------------------------
// GEMM: C[M,N] = A[M,K] @ Bw[K,N] + bias, 128x128x32 block tile, 8 waves,
// ping-pong LDS double buffering (1 barrier / k-step).
// MODE 0: QKV epilogue -> scatter f16 into Q/K/V [B,H,T,C], Q pre-scaled 1/8
// MODE 1: plain f32 store to Out[M,N]
// ---------------------------------------------------------------------------
template <typename AT, int MODE>
__global__ __launch_bounds__(256) void gemm128(
    const AT* __restrict__ A, const float* __restrict__ Bw,
    const float* __restrict__ bias,
    f16* __restrict__ Qb, f16* __restrict__ Kb, f16* __restrict__ Vb,
    float* __restrict__ Out, int N, int K) {
  const int tid  = threadIdx.x;
  const int lane = tid & 31, wave = tid >> 5;
  const int wm = wave >> 2, wn = wave & 3;      // 2 x 4 wave grid
  const int m0 = blockIdx.y * 128, n0 = blockIdx.x * 128;

  __shared__ f16 As[2][128][40];  // [buf][m][k], pad 8
  __shared__ f16 Bs[2][128][40];  // [buf][n][k] (transposed), pad 8

  const int r0 = tid >> 3, c4 = (tid & 7) * 4;   // A staging coords
  const int kk = tid >> 5, ng = (tid & 31) * 4;  // B staging coords

  AT    aReg[4][4];
  float bReg[4][4];

  auto loadTiles = [&](int kt) {
#pragma unroll
    for (int rr = 0; rr < 4; ++rr) {
      const AT* src = A + (size_t)(m0 + r0 + 32 * rr) * K + kt * 32 + c4;
#pragma unroll
      for (int j = 0; j < 4; ++j) aReg[rr][j] = src[j];
    }
#pragma unroll
    for (int i = 0; i < 4; ++i) {
      const float* src = Bw + (size_t)(kt * 32 + kk + 8 * i) * N + n0 + ng;
#pragma unroll
      for (int j = 0; j < 4; ++j) bReg[i][j] = src[j];
    }
  };
  auto storeTiles = [&](int buf) {
#pragma unroll
    for (int rr = 0; rr < 4; ++rr)
#pragma unroll
      for (int j = 0; j < 4; ++j)
        As[buf][r0 + 32 * rr][c4 + j] = (f16)(float)aReg[rr][j];
#pragma unroll
    for (int i = 0; i < 4; ++i)
#pragma unroll
      for (int j = 0; j < 4; ++j)
        Bs[buf][ng + j][kk + 8 * i] = (f16)bReg[i][j];
  };

  v8f acc[4][2];
#pragma unroll
  for (int mi = 0; mi < 4; ++mi)
#pragma unroll
    for (int ni = 0; ni < 2; ++ni) acc[mi][ni] = (v8f){0,0,0,0,0,0,0,0};

  const int kSteps = K / 32;
  loadTiles(0);
  storeTiles(0);
  __syncthreads();

  for (int kt = 0; kt < kSteps; ++kt) {
    const int cur = kt & 1;
    if (kt + 1 < kSteps) loadTiles(kt + 1);   // prefetch into regs

    v16h af[4], bf[2];
#pragma unroll
    for (int mi = 0; mi < 4; ++mi)
      af[mi] = load_a_frag(&As[cur][wm * 64 + mi * 16][0], 40);
#pragma unroll
    for (int ni = 0; ni < 2; ++ni)
      bf[ni] = load_b_frag(&Bs[cur][wn * 32 + ni * 16][0], 40);
#pragma unroll
    for (int mi = 0; mi < 4; ++mi)
#pragma unroll
      for (int ni = 0; ni < 2; ++ni)
        acc[mi][ni] = wmma_f32(af[mi], bf[ni], acc[mi][ni]);

    if (kt + 1 < kSteps) storeTiles(cur ^ 1);
    __syncthreads();
  }

  // ---- epilogue ----
  const int mlane = lane & 15, hf = lane >> 4;
#pragma unroll
  for (int mi = 0; mi < 4; ++mi) {
#pragma unroll
    for (int ni = 0; ni < 2; ++ni) {
#pragma unroll
      for (int r = 0; r < 8; ++r) {
        const int m = m0 + wm * 64 + mi * 16 + r + 8 * hf;
        const int n = n0 + wn * 32 + ni * 16 + mlane;
        const float v = acc[mi][ni][r] + bias[n];
        if (MODE == 0) {
          const int h = n / (3 * Cc), j = n % (3 * Cc);
          const int b = m >> 11, t = m & (Tt - 1);
          const size_t dst = ((size_t)(b * Hh + h) * Tt + t) * Cc;
          if (j < Cc)            Qb[dst + j]          = (f16)(v * SCALE2);
          else if (j < 2 * Cc)   Kb[dst + j - Cc]     = (f16)v;
          else                   Vb[dst + j - 2 * Cc] = (f16)v;
        } else {
          Out[(size_t)m * N + n] = v;
        }
      }
    }
  }
}

// ---------------------------------------------------------------------------
// k_mean[b,h,c] = mean_t K[b,h,t,c]
// ---------------------------------------------------------------------------
__global__ __launch_bounds__(64) void kmean_kernel(const f16* __restrict__ Kb,
                                                   float* __restrict__ kmean) {
  const int bh = blockIdx.x, c = threadIdx.x;
  const f16* Kp = Kb + (size_t)bh * Tt * Cc;
  float s = 0.f;
  for (int t = 0; t < Tt; ++t) s += (float)Kp[t * Cc + c];
  kmean[bh * Cc + c] = s * (1.0f / Tt);
}

// pos[b,t] = sum_{h,c} Qscaled[b,h,t,c] * k_mean[b,h,c]   (scale2 in Qscaled)
__global__ __launch_bounds__(128) void pos_kernel(const f16* __restrict__ Qb,
                                                  const float* __restrict__ kmean,
                                                  float* __restrict__ pos) {
  const int m = blockIdx.x, b = m >> 11, t = m & (Tt - 1);
  const int tid = threadIdx.x;
  float s = 0.f;
#pragma unroll
  for (int i = 0; i < 8; ++i) {
    const int idx = tid + i * 128;
    const int h = idx >> 6, c = idx & 63;
    s += (float)Qb[((size_t)(b * Hh + h) * Tt + t) * Cc + c] *
         kmean[(b * Hh + h) * Cc + c];
  }
  __shared__ float red[128];
  red[tid] = s;
  __syncthreads();
  for (int st = 64; st > 0; st >>= 1) {
    if (tid < st) red[tid] += red[tid + st];
    __syncthreads();
  }
  if (tid == 0) pos[m] = red[0];
}

// aw[b,t] = (pos - min) / (max - min + 1e-6)
__global__ __launch_bounds__(256) void aw_kernel(const float* __restrict__ pos,
                                                 float* __restrict__ aw) {
  const int b = blockIdx.x, tid = threadIdx.x;
  const float* p = pos + b * Tt;
  float mn = 1e30f, mx = -1e30f;
  for (int t = tid; t < Tt; t += 256) {
    const float v = p[t];
    mn = fminf(mn, v);
    mx = fmaxf(mx, v);
  }
  __shared__ float smn[256], smx[256];
  smn[tid] = mn; smx[tid] = mx;
  __syncthreads();
  for (int st = 128; st > 0; st >>= 1) {
    if (tid < st) {
      smn[tid] = fminf(smn[tid], smn[tid + st]);
      smx[tid] = fmaxf(smx[tid], smx[tid + st]);
    }
    __syncthreads();
  }
  const float gmn = smn[0];
  const float inv = 1.0f / (smx[0] - gmn + 1e-6f);
  for (int t = tid; t < Tt; t += 256) aw[b * Tt + t] = (p[t] - gmn) * inv;
}

// ---------------------------------------------------------------------------
// Flash attention per (b,h): 128 query rows/block, 64-key tiles, online
// softmax in fp32, output scaled by aw and written f16 to A2[B,T,W].
// Q/K tiles staged with GLOBAL_LOAD_ASYNC_TO_LDS (ASYNCcnt); V via VGPR
// transpose into B-operand layout.
// ---------------------------------------------------------------------------
__global__ __launch_bounds__(256) void attn_kernel(
    const f16* __restrict__ Qb, const f16* __restrict__ Kb,
    const f16* __restrict__ Vb, const float* __restrict__ aw,
    f16* __restrict__ A2) {
  const int tid = threadIdx.x;
  const int lane = tid & 31, wave = tid >> 5;
  const int mlane = lane & 15, hf = lane >> 4;
  const int bh = blockIdx.y, b = bh >> 4, h = bh & 15;
  const int m0t = blockIdx.x * 128;

  const f16* Qp = Qb + (size_t)bh * Tt * Cc;
  const f16* Kp = Kb + (size_t)bh * Tt * Cc;
  const f16* Vp = Vb + (size_t)bh * Tt * Cc;

  __shared__ f16 Qs[128][72];      // [row][c]
  __shared__ f16 Ks[64][72];       // [key][c]   (B operand layout for Q.K^T)
  __shared__ f16 Vs[64][72];       // [c][key]   (transposed: B operand for P.V)
  __shared__ f16 Ps[8][16][72];    // per-wave P round-trip (A operand layout)

  // stage Q tile once (async global->LDS, 64B per thread in 16B chunks)
  {
    const int row = tid >> 1, cg = (tid & 1) * 32;
    const uint32_t lds = lds_off_of(&Qs[row][cg]);
    const uint32_t goff =
        (uint32_t)(((uint32_t)(m0t + row) * Cc + cg) * sizeof(f16));
    const uint64_t base = (uint64_t)(uintptr_t)Qp;
#pragma unroll
    for (int i = 0; i < 4; ++i)
      async_copy_b128(lds + 16u * i, base, goff + 16u * i);
  }
  async_wait0();
  __syncthreads();

  v16h aQ[2];
#pragma unroll
  for (int cs = 0; cs < 2; ++cs)
    aQ[cs] = load_a_frag(&Qs[wave * 16][cs * 32], 72);

  float mrow[8], lrow[8];
#pragma unroll
  for (int r = 0; r < 8; ++r) { mrow[r] = -1e30f; lrow[r] = 0.f; }
  v8f ofrag[4];
#pragma unroll
  for (int ci = 0; ci < 4; ++ci) ofrag[ci] = (v8f){0,0,0,0,0,0,0,0};

  const int krow = tid >> 2, kcg = (tid & 3) * 16;  // staging coords
  const uint64_t kbase = (uint64_t)(uintptr_t)Kp;
  const uint32_t klds = lds_off_of(&Ks[krow][kcg]);

  for (int jb = 0; jb < Tt / 64; ++jb) {
    const int n0 = jb * 64;
    // K tile: async copy (natural [key][c] layout), 32B per thread
    {
      const uint32_t goff =
          (uint32_t)(((uint32_t)(n0 + krow) * Cc + kcg) * sizeof(f16));
      async_copy_b128(klds,       kbase, goff);
      async_copy_b128(klds + 16u, kbase, goff + 16u);
    }
    // V tile: VGPR load + transposed LDS store -> Vs[c][key]
    {
      const f16* src = Vp + (size_t)(n0 + krow) * Cc + kcg;
#pragma unroll
      for (int i = 0; i < 16; ++i) Vs[kcg + i][krow] = src[i];
    }
    async_wait0();
    __syncthreads();

    // S = Qscaled @ K^T   [16 rows x 64 keys per wave]
    v8f s[4];
#pragma unroll
    for (int ni = 0; ni < 4; ++ni) s[ni] = (v8f){0,0,0,0,0,0,0,0};
#pragma unroll
    for (int cs = 0; cs < 2; ++cs) {
#pragma unroll
      for (int ni = 0; ni < 4; ++ni) {
        v16h bk = load_b_frag(&Ks[ni * 16][cs * 32], 72);
        s[ni] = wmma_f32(aQ[cs], bk, s[ni]);
      }
    }

    // online softmax (rows live in 16-lane halves; reduce via shfl_xor 1..8)
    float fac[8];
#pragma unroll
    for (int r = 0; r < 8; ++r) {
      float vmax = -1e30f;
#pragma unroll
      for (int ni = 0; ni < 4; ++ni) vmax = fmaxf(vmax, s[ni][r]);
#pragma unroll
      for (int off = 1; off < 16; off <<= 1)
        vmax = fmaxf(vmax, __shfl_xor(vmax, off, 32));
      const float newm = fmaxf(mrow[r], vmax);
      float psum = 0.f;
#pragma unroll
      for (int ni = 0; ni < 4; ++ni) {
        const float p = __expf(s[ni][r] - newm);
        s[ni][r] = p;
        psum += p;
      }
#pragma unroll
      for (int off = 1; off < 16; off <<= 1)
        psum += __shfl_xor(psum, off, 32);
      fac[r] = __expf(mrow[r] - newm);
      lrow[r] = lrow[r] * fac[r] + psum;
      mrow[r] = newm;
    }
#pragma unroll
    for (int ci = 0; ci < 4; ++ci)
#pragma unroll
      for (int r = 0; r < 8; ++r) ofrag[ci][r] *= fac[r];

    // P -> LDS (A-operand layout source), then O += P @ V
#pragma unroll
    for (int ni = 0; ni < 4; ++ni)
#pragma unroll
      for (int r = 0; r < 8; ++r)
        Ps[wave][r + 8 * hf][ni * 16 + mlane] = (f16)s[ni][r];

#pragma unroll
    for (int cs = 0; cs < 2; ++cs) {
      v16h ap = load_a_frag(&Ps[wave][0][cs * 32], 72);
#pragma unroll
      for (int ci = 0; ci < 4; ++ci) {
        v16h bv = load_b_frag(&Vs[ci * 16][cs * 32], 72);
        ofrag[ci] = wmma_f32(ap, bv, ofrag[ci]);
      }
    }
    __syncthreads();
  }

  // epilogue: normalize, apply adaptive weight, store f16 [B,T,W]
  const float* awp = aw + b * Tt;
#pragma unroll
  for (int ci = 0; ci < 4; ++ci) {
#pragma unroll
    for (int r = 0; r < 8; ++r) {
      const int row = wave * 16 + r + 8 * hf;
      const int t = m0t + row;
      const float val = ofrag[ci][r] / lrow[r] * awp[t];
      const int c = ci * 16 + mlane;
      A2[(size_t)(b * Tt + t) * Ww + h * Cc + c] = (f16)val;
    }
  }
}

// ---------------------------------------------------------------------------
extern "C" void kernel_launch(void* const* d_in, const int* in_sizes, int n_in,
                              void* d_out, int out_size, void* d_ws,
                              size_t ws_size, hipStream_t stream) {
  const float* x      = (const float*)d_in[0];
  const float* W_qkv  = (const float*)d_in[1];
  const float* b_qkv  = (const float*)d_in[2];
  const float* W_proj = (const float*)d_in[3];
  const float* b_proj = (const float*)d_in[4];
  float* out = (float*)d_out;

  char* ws = (char*)d_ws;
  const size_t qsz = (size_t)Bc * Hh * Tt * Cc * sizeof(f16); // 8 MB
  f16* Qb = (f16*)(ws);
  f16* Kb = (f16*)(ws + qsz);
  f16* Vb = (f16*)(ws + 2 * qsz);
  f16* A2 = (f16*)(ws + 3 * qsz);
  float* kmean = (float*)(ws + 4 * qsz);
  float* pos   = (float*)(ws + 4 * qsz + 64 * 1024);
  float* aw    = (float*)(ws + 4 * qsz + 128 * 1024);

  const dim3 blk(256);

  // 1) QKV GEMM + bias + scatter to Q/K/V (f16, Q pre-scaled by 1/8)
  gemm128<float, 0><<<dim3(3 * Ww / 128, Bc * Tt / 128), blk, 0, stream>>>(
      x, W_qkv, b_qkv, Qb, Kb, Vb, nullptr, 3 * Ww, Ww);

  // 2) adaptive-weight path
  kmean_kernel<<<dim3(Bc * Hh), dim3(64), 0, stream>>>(Kb, kmean);
  pos_kernel<<<dim3(Bc * Tt), dim3(128), 0, stream>>>(Qb, kmean, pos);
  aw_kernel<<<dim3(Bc), dim3(256), 0, stream>>>(pos, aw);

  // 3) flash attention (aw fused) -> A2 f16 [B,T,W]
  attn_kernel<<<dim3(Tt / 128, Bc * Hh), blk, 0, stream>>>(Qb, Kb, Vb, aw, A2);

  // 4) projection GEMM + bias -> d_out f32
  gemm128<f16, 1><<<dim3(Ww / 128, Bc * Tt / 128), blk, 0, stream>>>(
      A2, W_proj, b_proj, nullptr, nullptr, nullptr, out, Ww, Ww);
}
